// HeteroRGCNLayer_88510686036237
// MI455X (gfx1250) — matrix operand order repaired and checked by reading, not compiled, not run on hardware
//
#include <hip/hip_runtime.h>

// ---------------- problem constants (match reference) ----------------
#define ND 100000
#define NG 100000
#define DOUT 512

typedef __attribute__((ext_vector_type(16))) __bf16 v16bf;
typedef __attribute__((ext_vector_type(8)))  float  v8f;

// native f32 -> bf16 convert (RNE); lowers to v_cvt_*bf16_f32 on gfx1250
__device__ __forceinline__ __bf16 f2bf(float f) { return (__bf16)f; }

__device__ __forceinline__ v16bf pack_frag(uint4 lo, uint4 hi) {
    unsigned tmp[8] = {lo.x, lo.y, lo.z, lo.w, hi.x, hi.y, hi.z, hi.w};
    v16bf r;
    __builtin_memcpy(&r, tmp, 32);
    return r;
}

// ---------------- small utility kernels ----------------
__global__ void zero_f4(float4* __restrict__ p, long long n4) {
    long long i = (long long)blockIdx.x * blockDim.x + threadIdx.x;
    if (i < n4) p[i] = make_float4(0.f, 0.f, 0.f, 0.f);
}

__global__ void degree_count(const int* __restrict__ src, const int* __restrict__ dst,
                             float* __restrict__ deg_out, float* __restrict__ deg_in, int E) {
    int e = blockIdx.x * blockDim.x + threadIdx.x;
    if (e < E) {
        atomicAdd(&deg_out[src[e]], 1.0f);
        atomicAdd(&deg_in[dst[e]], 1.0f);
    }
}

__global__ void deg_to_norm(float* __restrict__ d, int n) {
    int i = blockIdx.x * blockDim.x + threadIdx.x;
    if (i < n) d[i] = rsqrtf(fmaxf(d[i], 1.0f));
}

// msg = h_src[src]*norm_src[src]; agg[dst] += msg  (scatter-sum, float atomics)
// warp covers 32 consecutive float4 chunks of one edge -> coalesced 512B gathers
__global__ void scatter_edges(const float* __restrict__ h_src,
                              const int* __restrict__ src, const int* __restrict__ dst,
                              const float* __restrict__ norm_src,
                              float* __restrict__ agg, int E) {
    long long gid = (long long)blockIdx.x * blockDim.x + threadIdx.x;
    long long total = (long long)E * (DOUT / 4);
    if (gid >= total) return;
    int e = (int)(gid >> 7);          // /128
    int c = (int)(gid & 127);
    int s = src[e];
    int d = dst[e];
    float scale = norm_src[s];
    const float4 v = *(const float4*)&h_src[(size_t)s * DOUT + c * 4];
    float* dp = &agg[(size_t)d * DOUT + c * 4];
    atomicAdd(dp + 0, v.x * scale);
    atomicAdd(dp + 1, v.y * scale);
    atomicAdd(dp + 2, v.z * scale);
    atomicAdd(dp + 3, v.w * scale);
}

// ---------------- bf16 WMMA GEMM ----------------
// C[M,N] = alpha * (rowScale .* A[M,K]) @ W[K,N] + biasScale*bias[N]  (+= if accumulate)
// Block tile 128x128, K-step 32, 256 threads = 8 waves, wave tile 32x64.
// Register double-buffering: next K-step's global tiles are loaded while WMMAs
// run on the current LDS tile.
#define BM 128
#define BN 128
#define BK 32
#define LDSS 40  // padded row stride (bf16 elems) to spread LDS banks

__global__ __launch_bounds__(256)
void wmma_gemm_bf16(const float* __restrict__ A, const float* __restrict__ W,
                    const float* __restrict__ bias, const float* __restrict__ rowScale,
                    float* __restrict__ C, int M, int N, int K,
                    float alpha, float biasScale, int accumulate) {
    __shared__ __align__(16) __bf16 sA[BM * LDSS]; // [m][k]
    __shared__ __align__(16) __bf16 sB[BN * LDSS]; // [n][k] (W transposed)

    const int tid = threadIdx.x;
    const int lane = tid & 31;
    const int wave = tid >> 5;
    const int wm = wave >> 1;          // 0..3 -> 32-row strip
    const int wn = wave & 1;           // 0..1 -> 64-col strip
    const int rowBase = blockIdx.y * BM;
    const int colBase = blockIdx.x * BN;

    v8f acc[2][4];
#pragma unroll
    for (int i = 0; i < 2; ++i)
#pragma unroll
        for (int j = 0; j < 4; ++j) acc[i][j] = v8f{};

    const int mh = lane & 15;
    const int aK0 = (lane < 16) ? 0 : 8;    // A frag: K in {k0..k0+7, k0+16..k0+23}
    const int bK0 = (lane < 16) ? 0 : 8;    // B frag: K in {k0..k0+7, k0+16..k0+23}

    float4 pa[4]; float ps[4];              // staged A tile (+ per-row scale)
    float4 pb[4];                           // staged W tile

    auto loadA = [&](int kb) {
#pragma unroll
        for (int i = 0; i < 4; ++i) {
            int p = tid + i * 256;          // 0..1023 over 128 rows x 8 float4
            int r = p >> 3;
            int c4 = p & 7;
            int gr = rowBase + r;
            pa[i] = make_float4(0.f, 0.f, 0.f, 0.f);
            ps[i] = 0.f;
            if (gr < M) {
                const float* ap = &A[(size_t)gr * K + kb + c4 * 4];
                pa[i] = *(const float4*)ap;
                ps[i] = alpha * (rowScale ? rowScale[gr] : 1.0f);
                if (kb + 2 * BK < K) __builtin_prefetch(ap + 2 * BK, 0, 0);
            }
        }
    };
    auto loadB = [&](int kb) {
#pragma unroll
        for (int i = 0; i < 4; ++i) {
            int p = tid + i * 256;          // 0..1023 over 32 k-rows x 32 float4
            int r = p >> 5;
            int c4 = p & 31;
            pb[i] = *(const float4*)&W[(size_t)(kb + r) * N + colBase + c4 * 4];
        }
    };

    loadA(0);
    loadB(0);

    for (int kb = 0; kb < K; kb += BK) {
        // --- convert & store staged tiles to LDS (native bf16 cvt) ---
#pragma unroll
        for (int i = 0; i < 4; ++i) {
            int p = tid + i * 256;
            int r = p >> 3;
            int c4 = p & 7;
            __bf16* d = &sA[r * LDSS + c4 * 4];
            float s = ps[i];
            d[0] = f2bf(pa[i].x * s); d[1] = f2bf(pa[i].y * s);
            d[2] = f2bf(pa[i].z * s); d[3] = f2bf(pa[i].w * s);
        }
#pragma unroll
        for (int i = 0; i < 4; ++i) {
            int p = tid + i * 256;
            int r = p >> 5;
            int c4 = p & 31;
            int n0 = c4 * 4;
            sB[(n0 + 0) * LDSS + r] = f2bf(pb[i].x);
            sB[(n0 + 1) * LDSS + r] = f2bf(pb[i].y);
            sB[(n0 + 2) * LDSS + r] = f2bf(pb[i].z);
            sB[(n0 + 3) * LDSS + r] = f2bf(pb[i].w);
        }
        __syncthreads();

        // --- kick off next K-step's global loads (overlap with WMMA) ---
        if (kb + BK < K) {
            loadA(kb + BK);
            loadB(kb + BK);
        }

        // --- fragments per ISA layout + 8 WMMAs per wave ---
        v16bf aF[2], bF[4];
#pragma unroll
        for (int mt = 0; mt < 2; ++mt) {
            const __bf16* base = &sA[(wm * 32 + mt * 16 + mh) * LDSS];
            uint4 lo = *(const uint4*)(base + aK0);
            uint4 hi = *(const uint4*)(base + aK0 + 16);
            aF[mt] = pack_frag(lo, hi);
        }
#pragma unroll
        for (int nt = 0; nt < 4; ++nt) {
            const __bf16* base = &sB[(wn * 64 + nt * 16 + mh) * LDSS];
            uint4 lo = *(const uint4*)(base + bK0);
            uint4 hi = *(const uint4*)(base + bK0 + 16);
            bF[nt] = pack_frag(lo, hi);
        }
#pragma unroll
        for (int mt = 0; mt < 2; ++mt)
#pragma unroll
            for (int nt = 0; nt < 4; ++nt)
                acc[mt][nt] = __builtin_amdgcn_wmma_f32_16x16x32_bf16(
                    false, aF[mt], false, bF[nt], (short)0, acc[mt][nt], false, false);
        __syncthreads();
    }

    // --- epilogue: D layout = M: v + 8*(lane>=16), N: lane&15 ---
#pragma unroll
    for (int mt = 0; mt < 2; ++mt) {
#pragma unroll
        for (int nt = 0; nt < 4; ++nt) {
            int n = colBase + wn * 64 + nt * 16 + (lane & 15);
            float bn = bias ? bias[n] * biasScale : 0.f;
            int mBase = rowBase + wm * 32 + mt * 16 + ((lane >> 4) << 3);
#pragma unroll
            for (int v = 0; v < 8; ++v) {
                int m = mBase + v;
                if (m < M) {
                    size_t off = (size_t)m * N + n;
                    float val = acc[mt][nt][v] + bn;
                    if (accumulate) val += C[off];
                    C[off] = val;
                }
            }
        }
    }
}

// ---------------- launch ----------------
extern "C" void kernel_launch(void* const* d_in, const int* in_sizes, int n_in,
                              void* d_out, int out_size, void* d_ws, size_t ws_size,
                              hipStream_t stream) {
    const float* x_drug = (const float*)d_in[0];
    const float* x_gene = (const float*)d_in[1];
    const float* W_drug = (const float*)d_in[2];
    const float* b_drug = (const float*)d_in[3];
    const float* W_gene = (const float*)d_in[4];
    const float* b_gene = (const float*)d_in[5];
    const float* W_dd = (const float*)d_in[6];
    const float* b_dd = (const float*)d_in[7];
    const float* W_dg = (const float*)d_in[8];
    const float* b_dg = (const float*)d_in[9];
    const float* W_gd = (const float*)d_in[10];
    const float* b_gd = (const float*)d_in[11];
    const float* W_gg = (const float*)d_in[12];
    const float* b_gg = (const float*)d_in[13];
    const int* src_dd = (const int*)d_in[14];
    const int* dst_dd = (const int*)d_in[15];
    const int* src_dg = (const int*)d_in[16];
    const int* dst_dg = (const int*)d_in[17];
    const int* src_gd = (const int*)d_in[18];
    const int* dst_gd = (const int*)d_in[19];
    const int* src_gg = (const int*)d_in[20];
    const int* dst_gg = (const int*)d_in[21];
    const int E = in_sizes[14];

    const size_t feat = (size_t)ND * DOUT;          // 51.2M floats
    float* ws  = (float*)d_ws;
    float* h_d = ws;
    float* h_g = h_d + feat;
    float* agg = h_g + feat;
    float* deg = agg + feat;                        // 8 x 100000 floats
    float* no_dd = deg + 0 * ND; float* ni_dd = deg + 1 * ND;
    float* no_dg = deg + 2 * ND; float* ni_dg = deg + 3 * ND;
    float* no_gd = deg + 4 * ND; float* ni_gd = deg + 5 * ND;
    float* no_gg = deg + 6 * ND; float* ni_gg = deg + 7 * ND;

    float* out_drug = (float*)d_out;
    float* out_gene = out_drug + feat;

    // 1) degrees -> norms
    {
        long long n4 = (8LL * ND) / 4;
        zero_f4<<<(unsigned)((n4 + 255) / 256), 256, 0, stream>>>((float4*)deg, n4);
        int blk = (E + 255) / 256;
        degree_count<<<blk, 256, 0, stream>>>(src_dd, dst_dd, no_dd, ni_dd, E);
        degree_count<<<blk, 256, 0, stream>>>(src_dg, dst_dg, no_dg, ni_dg, E);
        degree_count<<<blk, 256, 0, stream>>>(src_gd, dst_gd, no_gd, ni_gd, E);
        degree_count<<<blk, 256, 0, stream>>>(src_gg, dst_gg, no_gg, ni_gg, E);
        deg_to_norm<<<(8 * ND + 255) / 256, 256, 0, stream>>>(deg, 8 * ND);
    }

    dim3 gGemm(DOUT / BN, (ND + BM - 1) / BM);

    // 2) per-ntype projections (bf16 WMMA, f32 accumulate)
    wmma_gemm_bf16<<<gGemm, 256, 0, stream>>>(x_drug, W_drug, b_drug, nullptr,
                                              h_d, ND, DOUT, 512, 1.f, 1.f, 0);
    wmma_gemm_bf16<<<gGemm, 256, 0, stream>>>(x_gene, W_gene, b_gene, nullptr,
                                              h_g, NG, DOUT, 256, 1.f, 1.f, 0);

    // 3) per-etype GraphConv: zero agg -> scatter(norm_src folded) ->
    //    GEMM(norm_dst folded as rowScale, alpha=0.5, bias*0.5), 2nd etype accumulates
    long long n4agg = (long long)feat / 4;
    unsigned zb = (unsigned)((n4agg + 255) / 256);
    long long sth = (long long)E * (DOUT / 4);
    unsigned sb = (unsigned)((sth + 255) / 256);

    // dd -> drug
    zero_f4<<<zb, 256, 0, stream>>>((float4*)agg, n4agg);
    scatter_edges<<<sb, 256, 0, stream>>>(h_d, src_dd, dst_dd, no_dd, agg, E);
    wmma_gemm_bf16<<<gGemm, 256, 0, stream>>>(agg, W_dd, b_dd, ni_dd,
                                              out_drug, ND, DOUT, DOUT, 0.5f, 0.5f, 0);
    // gd -> drug (accumulate)
    zero_f4<<<zb, 256, 0, stream>>>((float4*)agg, n4agg);
    scatter_edges<<<sb, 256, 0, stream>>>(h_g, src_gd, dst_gd, no_gd, agg, E);
    wmma_gemm_bf16<<<gGemm, 256, 0, stream>>>(agg, W_gd, b_gd, ni_gd,
                                              out_drug, ND, DOUT, DOUT, 0.5f, 0.5f, 1);
    // dg -> gene
    zero_f4<<<zb, 256, 0, stream>>>((float4*)agg, n4agg);
    scatter_edges<<<sb, 256, 0, stream>>>(h_d, src_dg, dst_dg, no_dg, agg, E);
    wmma_gemm_bf16<<<gGemm, 256, 0, stream>>>(agg, W_dg, b_dg, ni_dg,
                                              out_gene, NG, DOUT, DOUT, 0.5f, 0.5f, 0);
    // gg -> gene (accumulate)
    zero_f4<<<zb, 256, 0, stream>>>((float4*)agg, n4agg);
    scatter_edges<<<sb, 256, 0, stream>>>(h_g, src_gg, dst_gg, no_gg, agg, E);
    wmma_gemm_bf16<<<gGemm, 256, 0, stream>>>(agg, W_gg, b_gg, ni_gg,
                                              out_gene, NG, DOUT, DOUT, 0.5f, 0.5f, 1);
}
